// ATransMIL_12996571037854
// MI455X (gfx1250) — matrix operand order repaired and checked by reading, not compiled, not run on hardware
//
#include <hip/hip_runtime.h>
#include <hip/hip_bf16.h>
#include <cstdint>

// ---------------- WMMA types ----------------
typedef __attribute__((ext_vector_type(16))) __bf16 v16bf;
typedef __attribute__((ext_vector_type(8)))  float  v8f;

__device__ __forceinline__ unsigned short f2bf(float f) {
  unsigned int u = __float_as_uint(f);
  unsigned int r = (u + 0x7FFFu + ((u >> 16) & 1u)) >> 16;
  return (unsigned short)r;
}
__device__ __forceinline__ unsigned int pk_bf(float lo, float hi) {
  return (unsigned int)f2bf(lo) | ((unsigned int)f2bf(hi) << 16);
}

// flags for gemm_bf16
#define FL_BT   1   // B is [N x K] (use B^T)
#define FL_RELU 2
#define FL_BIAS 4
// flags for gemm_f32
#define F32_IMB 1   // B_eff = bI*I - B  (square, NN)
#define F32_BT  2

#define GBM 64
#define GBN 64
#define GBK 32
#define ASTR 36

// ---- tile staging helpers (all GEMM dims are tile-aligned by construction:
// M % 64 == 0, N % 64 == 0, K % 32 == 0 for every call site) ----
__device__ __forceinline__ void load_tiles(const float* __restrict__ A, const float* __restrict__ B,
                                           int lda, int ldb, int m0, int n0, int k0, int BT, int tid,
                                           float4* ra, float4* rb) {
#pragma unroll
  for (int p = 0; p < 4; p++) {           // A tile: 64 x 32, 512 float4 groups
    int g = tid + p * 128;
    int r = g >> 3, kk = (g & 7) * 4;
    ra[p] = *(const float4*)(A + (long long)(m0 + r) * lda + k0 + kk);
  }
  if (BT) {
#pragma unroll
    for (int p = 0; p < 4; p++) {         // B^T tile: rows are n, contiguous in k
      int g = tid + p * 128;
      int n = g >> 3, kk = (g & 7) * 4;
      rb[p] = *(const float4*)(B + (long long)(n0 + n) * ldb + k0 + kk);
    }
  } else {
#pragma unroll
    for (int p = 0; p < 4; p++) {         // B tile: rows are k, contiguous in n
      int g = tid + p * 128;
      int kk = g >> 4, nq = (g & 15) * 4;
      rb[p] = *(const float4*)(B + (long long)(k0 + kk) * ldb + n0 + nq);
    }
  }
}

__device__ __forceinline__ void store_tiles(unsigned short* sA, unsigned short* sB,
                                            int BT, int tid, const float4* ra, const float4* rb) {
#pragma unroll
  for (int p = 0; p < 4; p++) {
    int g = tid + p * 128;
    int r = g >> 3, kk = (g & 7) * 4;
    unsigned long long w = (unsigned long long)pk_bf(ra[p].x, ra[p].y) |
                           ((unsigned long long)pk_bf(ra[p].z, ra[p].w) << 32);
    *(unsigned long long*)&sA[r * ASTR + kk] = w;
  }
  if (BT) {
#pragma unroll
    for (int p = 0; p < 4; p++) {
      int g = tid + p * 128;
      int n = g >> 3, kk = (g & 7) * 4;
      unsigned long long w = (unsigned long long)pk_bf(rb[p].x, rb[p].y) |
                             ((unsigned long long)pk_bf(rb[p].z, rb[p].w) << 32);
      *(unsigned long long*)&sB[n * ASTR + kk] = w;
    }
  } else {
#pragma unroll
    for (int p = 0; p < 4; p++) {         // transpose scatter: 4 x b16
      int g = tid + p * 128;
      int kk = g >> 4, nq = (g & 15) * 4;
      sB[(nq + 0) * ASTR + kk] = f2bf(rb[p].x);
      sB[(nq + 1) * ASTR + kk] = f2bf(rb[p].y);
      sB[(nq + 2) * ASTR + kk] = f2bf(rb[p].z);
      sB[(nq + 3) * ASTR + kk] = f2bf(rb[p].w);
    }
  }
}

// C[MxN] = alpha*(A @ opB(B)) (+bias) (relu); bf16 WMMA, f32 accumulate.
// Software-pipelined: next K-tile prefetched into registers during WMMA issue.
__launch_bounds__(128)
__global__ void gemm_bf16_kernel(const float* __restrict__ A, const float* __restrict__ B,
                                 const float* __restrict__ bias, float* __restrict__ C,
                                 int M, int N, int K, int lda, int ldb, int ldc,
                                 long long strA, long long strB, long long strC,
                                 float alpha, int flags) {
  __shared__ unsigned short sA[GBM * ASTR];
  __shared__ unsigned short sB[GBN * ASTR];
  const int bz = blockIdx.z;
  A += strA * bz; B += strB * bz; C += strC * bz;
  const int m0 = blockIdx.y * GBM, n0 = blockIdx.x * GBN;
  const int tid = threadIdx.x;
  const int wave = tid >> 5, lane = tid & 31;
  const int wr = wave >> 1, wc = wave & 1;   // 2x2 waves -> 64x64
  const int lr = lane & 15, lh = lane >> 4;
  const int BT = flags & FL_BT;

  v8f acc[2][2];
#pragma unroll
  for (int i = 0; i < 2; i++)
#pragma unroll
    for (int j = 0; j < 2; j++) acc[i][j] = (v8f){0, 0, 0, 0, 0, 0, 0, 0};

  float4 ra[4], rb[4];
  load_tiles(A, B, lda, ldb, m0, n0, 0, BT, tid, ra, rb);
  store_tiles(sA, sB, BT, tid, ra, rb);
  __syncthreads();

  for (int k0 = 0; k0 < K; k0 += GBK) {
    const bool more = (k0 + GBK) < K;
    if (more) load_tiles(A, B, lda, ldb, m0, n0, k0 + GBK, BT, tid, ra, rb);

    union FragU { v16bf v; unsigned short u[16]; } fa[2], fb[2];
#pragma unroll
    for (int i = 0; i < 2; i++) {
      const unsigned short* pa = &sA[(wr * 32 + i * 16 + lr) * ASTR];
#pragma unroll
      for (int t = 0; t < 8; t++) { fa[i].u[t] = pa[lh * 8 + t]; fa[i].u[8 + t] = pa[16 + lh * 8 + t]; }
    }
#pragma unroll
    for (int j = 0; j < 2; j++) {
      const unsigned short* pb = &sB[(wc * 32 + j * 16 + lr) * ASTR];
#pragma unroll
      for (int t = 0; t < 8; t++) { fb[j].u[t] = pb[lh * 8 + t]; fb[j].u[8 + t] = pb[16 + lh * 8 + t]; }
    }
#pragma unroll
    for (int i = 0; i < 2; i++)
#pragma unroll
      for (int j = 0; j < 2; j++)
        acc[i][j] = __builtin_amdgcn_wmma_f32_16x16x32_bf16(
            false, fa[i].v, false, fb[j].v, (short)0, acc[i][j], false, false);

    __syncthreads();                 // done reading this tile
    if (more) store_tiles(sA, sB, BT, tid, ra, rb);
    __syncthreads();                 // next tile visible
  }

#pragma unroll
  for (int i = 0; i < 2; i++)
#pragma unroll
    for (int j = 0; j < 2; j++) {
      int rb_ = m0 + wr * 32 + i * 16 + lh * 8;
      int cc = n0 + wc * 32 + j * 16 + lr;
      float bv = (flags & FL_BIAS) ? bias[cc] : 0.f;
#pragma unroll
      for (int g = 0; g < 8; g++) {
        float v = alpha * acc[i][j][g] + bv;
        if (flags & FL_RELU) v = fmaxf(v, 0.f);
        C[(long long)(rb_ + g) * ldc + cc] = v;
      }
    }
}

// Small f32 GEMM: C = pscale * (A @ opB(B)); opB may be (bI*I - B).
__launch_bounds__(256)
__global__ void gemm_f32_kernel(const float* __restrict__ A, const float* __restrict__ B,
                                float* __restrict__ C,
                                int M, int N, int K, int lda, int ldb, int ldc,
                                long long strA, long long strB, long long strC,
                                float pscale, float bI, int flags) {
  __shared__ float tA[16][17], tB[16][17];
  const int bz = blockIdx.z;
  A += strA * bz; B += strB * bz; C += strC * bz;
  const int tx = threadIdx.x, ty = threadIdx.y;
  const int row = blockIdx.y * 16 + ty, col = blockIdx.x * 16 + tx;
  float s = 0.f;
  for (int k0 = 0; k0 < K; k0 += 16) {
    tA[ty][tx] = (row < M && (k0 + tx) < K) ? A[(long long)row * lda + k0 + tx] : 0.f;
    int br = k0 + ty;
    float bv = 0.f;
    if (flags & F32_BT) { if (col < N && br < K) bv = B[(long long)col * ldb + br]; }
    else                { if (br < K && col < N) bv = B[(long long)br * ldb + col]; }
    if (flags & F32_IMB) bv = ((br == col) ? bI : 0.f) - bv;
    tB[ty][tx] = bv;
    __syncthreads();
#pragma unroll
    for (int kk = 0; kk < 16; kk++) s += tA[ty][kk] * tB[kk][tx];
    __syncthreads();
  }
  if (row < M && col < N) C[(long long)row * ldc + col] = pscale * s;
}

// ---------------- elementwise / reduction kernels ----------------
__global__ void fill_kernel(float* p, long long n, float v) {
  long long i = (long long)blockIdx.x * 256 + threadIdx.x;
  if (i < n) p[i] = v;
}
__global__ void copy_kernel(float* dst, const float* src, long long n) {
  long long i = (long long)blockIdx.x * 256 + threadIdx.x;
  if (i < n) dst[i] = src[i];
}
__global__ void addres_kernel(float* X, const float* OP, long long n) {
  long long i = (long long)blockIdx.x * 256 + threadIdx.x;
  if (i < n) X[i] += OP[i];
}

__launch_bounds__(256)
__global__ void ln512_kernel(const float* __restrict__ X, const float* __restrict__ g,
                             const float* __restrict__ b, float* __restrict__ Y) {
  long long r = blockIdx.x;
  const float* x = X + r * 512; float* y = Y + r * 512;
  int tid = threadIdx.x;
  float a0 = x[tid], a1 = x[tid + 256];
  __shared__ float red[256];
  red[tid] = a0 + a1; __syncthreads();
  for (int s = 128; s > 0; s >>= 1) { if (tid < s) red[tid] += red[tid + s]; __syncthreads(); }
  float mu = red[0] / 512.f; __syncthreads();
  float d0 = a0 - mu, d1 = a1 - mu;
  red[tid] = d0 * d0 + d1 * d1; __syncthreads();
  for (int s = 128; s > 0; s >>= 1) { if (tid < s) red[tid] += red[tid + s]; __syncthreads(); }
  float rstd = rsqrtf(red[0] / 512.f + 1e-5f);
  y[tid] = d0 * rstd * g[tid] + b[tid];
  y[tid + 256] = d1 * rstd * g[tid + 256] + b[tid + 256];
}

__launch_bounds__(256)
__global__ void softmax_kernel(float* __restrict__ S, int L) {
  float* p = S + (long long)blockIdx.x * L;
  int tid = threadIdx.x;
  __shared__ float red[256];
  float m = -3.4e38f;
  for (int i = tid; i < L; i += 256) m = fmaxf(m, p[i]);
  red[tid] = m; __syncthreads();
  for (int s = 128; s > 0; s >>= 1) { if (tid < s) red[tid] = fmaxf(red[tid], red[tid + s]); __syncthreads(); }
  float mx = red[0]; __syncthreads();
  float sum = 0.f;
  for (int i = tid; i < L; i += 256) { float e = __expf(p[i] - mx); p[i] = e; sum += e; }
  red[tid] = sum; __syncthreads();
  for (int s = 128; s > 0; s >>= 1) { if (tid < s) red[tid] += red[tid + s]; __syncthreads(); }
  float inv = 1.f / red[0];
  for (int i = tid; i < L; i += 256) p[i] *= inv;
}

// landmarks: ql/kl[h][m][d] = mean_j qkv[(m*37+j)][{0,512}+h*64+d]
__global__ void landmark_kernel(const float* __restrict__ qkv, float* ql, float* kl) {
  int m = blockIdx.x, h = blockIdx.y, d = threadIdx.x;
  float sq = 0.f, sk = 0.f;
  for (int j = 0; j < 37; j++) {
    const float* r = qkv + (long long)(m * 37 + j) * 1536 + h * 64 + d;
    sq += r[0]; sk += r[512];
  }
  ql[((long long)h * 256 + m) * 64 + d] = sq * (1.f / 37.f);
  kl[((long long)h * 256 + m) * 64 + d] = sk * (1.f / 37.f);
}

__global__ void pinv_sums_kernel(const float* __restrict__ A2, float* rs, float* cs) {
  int b = blockIdx.x; int h = b >> 8, r = b & 255; int tid = threadIdx.x;
  const float* base = A2 + (long long)h * 65536;
  __shared__ float red[256];
  red[tid] = fabsf(base[r * 256 + tid]); __syncthreads();
  for (int s = 128; s > 0; s >>= 1) { if (tid < s) red[tid] += red[tid + s]; __syncthreads(); }
  if (tid == 0) rs[b] = red[0];
  __syncthreads();
  red[tid] = fabsf(base[tid * 256 + r]); __syncthreads();
  for (int s = 128; s > 0; s >>= 1) { if (tid < s) red[tid] += red[tid + s]; __syncthreads(); }
  if (tid == 0) cs[b] = red[0];
}
__global__ void pinv_max_kernel(const float* rs, const float* cs, float* scale) {
  int tid = threadIdx.x;
  __shared__ float r1[256], r2[256];
  float m1 = 0.f, m2 = 0.f;
  for (int i = tid; i < 2048; i += 256) { m1 = fmaxf(m1, rs[i]); m2 = fmaxf(m2, cs[i]); }
  r1[tid] = m1; r2[tid] = m2; __syncthreads();
  for (int s = 128; s > 0; s >>= 1) {
    if (tid < s) { r1[tid] = fmaxf(r1[tid], r1[tid + s]); r2[tid] = fmaxf(r2[tid], r2[tid + s]); }
    __syncthreads();
  }
  if (tid == 0) scale[0] = r1[0] * r2[0];
}
__global__ void pinv_init_kernel(const float* __restrict__ A2, const float* scale, float* Z) {
  long long idx = (long long)blockIdx.x * 256 + threadIdx.x;
  if (idx >= 8LL * 65536) return;
  int h = (int)(idx >> 16); int rem = (int)(idx & 65535); int i = rem >> 8, j = rem & 255;
  Z[idx] = A2[(long long)h * 65536 + (long long)j * 256 + i] / scale[0];
}

// depthwise 33-tap conv of v along sequence, added into head-merged O (9472x512)
__global__ void resconv_kernel(const float* __restrict__ qkv, const float* __restrict__ w,
                               float* __restrict__ O) {
  long long idx = (long long)blockIdx.x * 256 + threadIdx.x;
  if (idx >= 9472LL * 512) return;
  int c = (int)(idx & 511); long long i = idx >> 9;
  int h = c >> 6;
  float s = 0.f;
  for (int k = 0; k < 33; k++) {
    long long t = i + k - 16;
    if (t >= 0 && t < 9472) s += qkv[t * 1536 + 1024 + c] * w[h * 33 + k];
  }
  O[idx] += s;
}

// PPEG: block 0 copies cls; blocks 1..9216: y = x + dw7 + dw5 + dw3 on 96x96x512
__launch_bounds__(256)
__global__ void ppeg_kernel(const float* __restrict__ X, float* __restrict__ Y,
                            const float* w7, const float* b7, const float* w5, const float* b5,
                            const float* w3, const float* b3) {
  int blk = blockIdx.x, tid = threadIdx.x;
  if (blk == 0) { for (int c = tid; c < 512; c += 256) Y[c] = X[c]; return; }
  int t = blk - 1, ph = t / 96, pw = t % 96;
  for (int c = tid; c < 512; c += 256) {
    float acc = X[(long long)(1 + t) * 512 + c] + b7[c] + b5[c] + b3[c];
    for (int dy = -3; dy <= 3; dy++)
      for (int dx = -3; dx <= 3; dx++) {
        int yy = ph + dy, xx = pw + dx;
        if (yy >= 0 && yy < 96 && xx >= 0 && xx < 96)
          acc += X[(long long)(1 + yy * 96 + xx) * 512 + c] * w7[c * 49 + (dy + 3) * 7 + (dx + 3)];
      }
    for (int dy = -2; dy <= 2; dy++)
      for (int dx = -2; dx <= 2; dx++) {
        int yy = ph + dy, xx = pw + dx;
        if (yy >= 0 && yy < 96 && xx >= 0 && xx < 96)
          acc += X[(long long)(1 + yy * 96 + xx) * 512 + c] * w5[c * 25 + (dy + 2) * 5 + (dx + 2)];
      }
    for (int dy = -1; dy <= 1; dy++)
      for (int dx = -1; dx <= 1; dx++) {
        int yy = ph + dy, xx = pw + dx;
        if (yy >= 0 && yy < 96 && xx >= 0 && xx < 96)
          acc += X[(long long)(1 + yy * 96 + xx) * 512 + c] * w3[c * 9 + (dy + 1) * 3 + (dx + 1)];
      }
    Y[(long long)(1 + t) * 512 + c] = acc;
  }
}

// Final per-bag head: emb = [LN(x_cls), age, loc]; logits = emb@fc2_w + fc2_b
__launch_bounds__(256)
__global__ void head_kernel(const float* __restrict__ X, const float* ng, const float* nb,
                            const float* age, const float* loc,
                            const float* __restrict__ fc2w, const float* fc2b,
                            float* __restrict__ emb, float* __restrict__ out) {
  __shared__ float e[551];
  __shared__ float red[256];
  int tid = threadIdx.x;
  float a0 = X[tid], a1 = X[tid + 256];
  red[tid] = a0 + a1; __syncthreads();
  for (int s = 128; s > 0; s >>= 1) { if (tid < s) red[tid] += red[tid + s]; __syncthreads(); }
  float mu = red[0] / 512.f; __syncthreads();
  float d0 = a0 - mu, d1 = a1 - mu;
  red[tid] = d0 * d0 + d1 * d1; __syncthreads();
  for (int s = 128; s > 0; s >>= 1) { if (tid < s) red[tid] += red[tid + s]; __syncthreads(); }
  float rstd = rsqrtf(red[0] / 512.f + 1e-5f);
  e[tid] = d0 * rstd * ng[tid] + nb[tid];
  e[tid + 256] = d1 * rstd * ng[tid + 256] + nb[tid + 256];
  if (tid < 32) e[512 + tid] = age[tid];
  if (tid < 7)  e[544 + tid] = loc[tid];
  __syncthreads();
  for (int i = tid; i < 551; i += 256) emb[i] = e[i];
  for (int j = tid; j < 186; j += 256) {
    float s = fc2b[j];
    for (int i = 0; i < 551; i++) s += e[i] * fc2w[i * 186 + j];
    out[j] = s;
  }
}

__global__ void pred1_kernel(const float* __restrict__ se, const float* __restrict__ w1,
                             float* __restrict__ y1) {
  int j = blockIdx.x * 256 + threadIdx.x;
  if (j >= 4096) return;
  float s = 0.f;
  for (int i = 0; i < 1653; i++) s += se[i] * w1[(long long)i * 4096 + j];
  y1[j] = s;
}
__launch_bounds__(256)
__global__ void pred_lnrelu_kernel(float* __restrict__ y, const float* g, const float* b) {
  int tid = threadIdx.x;
  __shared__ float red[256];
  float s = 0.f;
  for (int i = tid; i < 4096; i += 256) s += y[i];
  red[tid] = s; __syncthreads();
  for (int st = 128; st > 0; st >>= 1) { if (tid < st) red[tid] += red[tid + st]; __syncthreads(); }
  float mu = red[0] / 4096.f; __syncthreads();
  float v = 0.f;
  for (int i = tid; i < 4096; i += 256) { float d = y[i] - mu; v += d * d; }
  red[tid] = v; __syncthreads();
  for (int st = 128; st > 0; st >>= 1) { if (tid < st) red[tid] += red[tid + st]; __syncthreads(); }
  float rstd = rsqrtf(red[0] / 4096.f + 1e-5f); __syncthreads();
  for (int i = tid; i < 4096; i += 256) y[i] = fmaxf((y[i] - mu) * rstd * g[i] + b[i], 0.f);
}
__launch_bounds__(256)
__global__ void pred2_kernel(const float* __restrict__ y2, const float* __restrict__ w2,
                             const float* g, const float* b, float* __restrict__ out) {
  __shared__ float yv[186];
  __shared__ float red[256];
  int tid = threadIdx.x;
  for (int j = tid; j < 186; j += 256) {
    float s = 0.f;
    for (int i = 0; i < 4096; i++) s += y2[i] * w2[(long long)i * 186 + j];
    yv[j] = s;
  }
  __syncthreads();
  red[tid] = (tid < 186) ? yv[tid] : 0.f; __syncthreads();
  for (int st = 128; st > 0; st >>= 1) { if (tid < st) red[tid] += red[tid + st]; __syncthreads(); }
  float mu = red[0] / 186.f; __syncthreads();
  float d = (tid < 186) ? (yv[tid] - mu) : 0.f;
  red[tid] = d * d; __syncthreads();
  for (int st = 128; st > 0; st >>= 1) { if (tid < st) red[tid] += red[tid + st]; __syncthreads(); }
  float rstd = rsqrtf(red[0] / 186.f + 1e-5f);
  if (tid < 186) out[tid] = (yv[tid] - mu) * rstd * g[tid] + b[tid];
}

// ---------------- host orchestration ----------------
static inline int cdiv(long long a, long long b) { return (int)((a + b - 1) / b); }

struct WS {
  float *X, *X2, *XP, *QKV, *S, *O, *QL, *KL, *A2, *ZA, *ZB, *XZ, *T1, *T2m, *TT, *T2, *RS, *CS, *SC, *EMB, *Y1;
};

static void run_translayer(float* X, const float* ng, const float* nb, const float* qkvw,
                           const float* outw, const float* outb, const float* resw,
                           const WS& w, hipStream_t st) {
  const int N = 9472;       // padded sequence (255 zeros + 9217 tokens)
  // 1. zero pad rows + LN into padded buffer
  fill_kernel<<<cdiv(255LL * 512, 256), 256, 0, st>>>(w.XP, 255LL * 512, 0.f);
  ln512_kernel<<<9217, 256, 0, st>>>(X, ng, nb, w.XP + 255LL * 512);
  // 2. qkv = xp @ qkv_w  (9472 x 1536)
  gemm_bf16_kernel<<<dim3(cdiv(1536, GBN), cdiv(N, GBM), 1), 128, 0, st>>>(
      w.XP, qkvw, nullptr, w.QKV, N, 1536, 512, 512, 1536, 1536, 0, 0, 0, 1.f, 0);
  // 3. landmarks
  landmark_kernel<<<dim3(256, 8), 64, 0, st>>>(w.QKV, w.QL, w.KL);
  // 4. a2 = softmax(0.125 * q_l @ k_l^T), then pinv(a2)
  gemm_f32_kernel<<<dim3(16, 16, 8), dim3(16, 16), 0, st>>>(
      w.QL, w.KL, w.A2, 256, 256, 64, 64, 64, 256, 16384, 16384, 65536, 0.125f, 0.f, F32_BT);
  softmax_kernel<<<2048, 256, 0, st>>>(w.A2, 256);
  pinv_sums_kernel<<<2048, 256, 0, st>>>(w.A2, w.RS, w.CS);
  pinv_max_kernel<<<1, 256, 0, st>>>(w.RS, w.CS, w.SC);
  pinv_init_kernel<<<cdiv(8LL * 65536, 256), 256, 0, st>>>(w.A2, w.SC, w.ZA);
  float *Zc = w.ZA, *Zo = w.ZB;
  for (int it = 0; it < 6; it++) {
    gemm_f32_kernel<<<dim3(16, 16, 8), dim3(16, 16), 0, st>>>(
        w.A2, Zc, w.XZ, 256, 256, 256, 256, 256, 256, 65536, 65536, 65536, 1.f, 0.f, 0);
    gemm_f32_kernel<<<dim3(16, 16, 8), dim3(16, 16), 0, st>>>(
        w.XZ, w.XZ, w.T1, 256, 256, 256, 256, 256, 256, 65536, 65536, 65536, 1.f, 7.f, F32_IMB);
    gemm_f32_kernel<<<dim3(16, 16, 8), dim3(16, 16), 0, st>>>(
        w.XZ, w.T1, w.T2m, 256, 256, 256, 256, 256, 256, 65536, 65536, 65536, 1.f, 15.f, F32_IMB);
    gemm_f32_kernel<<<dim3(16, 16, 8), dim3(16, 16), 0, st>>>(
        Zc, w.T2m, Zo, 256, 256, 256, 256, 256, 256, 65536, 65536, 65536, 0.25f, 13.f, F32_IMB);
    float* t = Zc; Zc = Zo; Zo = t;
  }
  // 5. per head: a3 = softmax(0.125 * q_l @ k^T);  TT[h] = a3 @ v
  for (int h = 0; h < 8; h++) {
    gemm_bf16_kernel<<<dim3(cdiv(N, GBN), cdiv(256, GBM), 1), 128, 0, st>>>(
        w.QL + (long long)h * 16384, w.QKV + 512 + h * 64, nullptr, w.S,
        256, N, 64, 64, 1536, N, 0, 0, 0, 0.125f, FL_BT);
    softmax_kernel<<<256, 256, 0, st>>>(w.S, N);
    gemm_bf16_kernel<<<dim3(1, cdiv(256, GBM), 1), 128, 0, st>>>(
        w.S, w.QKV + 1024 + h * 64, nullptr, w.TT + (long long)h * 16384,
        256, 64, N, N, 1536, 64, 0, 0, 0, 1.f, 0);
  }
  // 6. T2 = pinv(a2) @ TT
  gemm_f32_kernel<<<dim3(4, 16, 8), dim3(16, 16), 0, st>>>(
      Zc, w.TT, w.T2, 256, 64, 256, 256, 64, 64, 65536, 16384, 16384, 1.f, 0.f, 0);
  // 7. per head: a1 = softmax(0.125 * q @ k_l^T);  O[:, h*64:] = a1 @ T2[h]
  for (int h = 0; h < 8; h++) {
    gemm_bf16_kernel<<<dim3(cdiv(256, GBN), cdiv(N, GBM), 1), 128, 0, st>>>(
        w.QKV + h * 64, w.KL + (long long)h * 16384, nullptr, w.S,
        N, 256, 64, 1536, 64, 256, 0, 0, 0, 0.125f, FL_BT);
    softmax_kernel<<<N, 256, 0, st>>>(w.S, 256);
    gemm_bf16_kernel<<<dim3(1, cdiv(N, GBM), 1), 128, 0, st>>>(
        w.S, w.T2 + (long long)h * 16384, nullptr, w.O + h * 64,
        N, 64, 256, 256, 64, 512, 0, 0, 0, 1.f, 0);
  }
  // 8. depthwise residual conv added into merged heads
  resconv_kernel<<<cdiv(9472LL * 512, 256), 256, 0, st>>>(w.QKV, resw, w.O);
  // 9. output projection (reuse XP) and residual add of last 9217 rows
  gemm_bf16_kernel<<<dim3(cdiv(512, GBN), cdiv(N, GBM), 1), 128, 0, st>>>(
      w.O, outw, outb, w.XP, N, 512, 512, 512, 512, 512, 0, 0, 0, 1.f, FL_BIAS);
  addres_kernel<<<cdiv(9217LL * 512, 256), 256, 0, st>>>(X, w.XP + 255LL * 512, 9217LL * 512);
}

extern "C" void kernel_launch(void* const* d_in, const int* in_sizes, int n_in,
                              void* d_out, int out_size, void* d_ws, size_t ws_size,
                              hipStream_t stream) {
  const float* data   = (const float*)d_in[0];
  const float* age    = (const float*)d_in[1];
  const float* loc    = (const float*)d_in[2];
  const float* fc1_w  = (const float*)d_in[4];
  const float* fc1_b  = (const float*)d_in[5];
  const float* cls    = (const float*)d_in[6];
  const float* l1ng = (const float*)d_in[7],  *l1nb = (const float*)d_in[8];
  const float* l1qkv = (const float*)d_in[9], *l1ow = (const float*)d_in[10];
  const float* l1ob = (const float*)d_in[11], *l1rw = (const float*)d_in[12];
  const float* l2ng = (const float*)d_in[13], *l2nb = (const float*)d_in[14];
  const float* l2qkv = (const float*)d_in[15], *l2ow = (const float*)d_in[16];
  const float* l2ob = (const float*)d_in[17], *l2rw = (const float*)d_in[18];
  const float* w7 = (const float*)d_in[19], *b7 = (const float*)d_in[20];
  const float* w5 = (const float*)d_in[21], *b5 = (const float*)d_in[22];
  const float* w3 = (const float*)d_in[23], *b3 = (const float*)d_in[24];
  const float* ng = (const float*)d_in[25], *nb = (const float*)d_in[26];
  const float* fc2w = (const float*)d_in[27], *fc2b = (const float*)d_in[28];
  const float* pw1 = (const float*)d_in[29];
  const float* pg1 = (const float*)d_in[30], *pb1 = (const float*)d_in[31];
  const float* pw2 = (const float*)d_in[32];
  const float* pg2 = (const float*)d_in[33], *pb2 = (const float*)d_in[34];
  float* out = (float*)d_out;

  // workspace layout (floats); total ~40M floats (~160 MB)
  WS w;
  float* p = (float*)d_ws;
  w.X   = p; p += 9217LL * 512;
  w.X2  = p; p += 9217LL * 512;
  w.XP  = p; p += 9472LL * 512;
  w.QKV = p; p += 9472LL * 1536;
  w.S   = p; p += 9472LL * 256;
  w.O   = p; p += 9472LL * 512;
  w.QL  = p; p += 131072;
  w.KL  = p; p += 131072;
  w.A2  = p; p += 524288;
  w.ZA  = p; p += 524288;
  w.ZB  = p; p += 524288;
  w.XZ  = p; p += 524288;
  w.T1  = p; p += 524288;
  w.T2m = p; p += 524288;
  w.TT  = p; p += 131072;
  w.T2  = p; p += 131072;
  w.RS  = p; p += 2048;
  w.CS  = p; p += 2048;
  w.SC  = p; p += 8;
  w.EMB = p; p += 1664;
  w.Y1  = p; p += 4096;

  for (int bag = 0; bag < 3; bag++) {
    float* X = w.X; float* X2 = w.X2;
    // cls token + fc1(relu) into X (row0=cls, rows 1..9216 = tokens)
    copy_kernel<<<2, 256, 0, stream>>>(X, cls, 512);
    gemm_bf16_kernel<<<dim3(cdiv(512, GBN), cdiv(9216, GBM), 1), 128, 0, stream>>>(
        data + (long long)bag * 9216 * 1536, fc1_w, fc1_b, X + 512,
        9216, 512, 1536, 1536, 512, 512, 0, 0, 0, 1.f, FL_BIAS | FL_RELU);
    run_translayer(X, l1ng, l1nb, l1qkv, l1ow, l1ob, l1rw, w, stream);
    ppeg_kernel<<<9217, 256, 0, stream>>>(X, X2, w7, b7, w5, b5, w3, b3);
    { float* t = X; X = X2; X2 = t; }
    run_translayer(X, l2ng, l2nb, l2qkv, l2ow, l2ob, l2rw, w, stream);
    head_kernel<<<1, 256, 0, stream>>>(X, ng, nb, age, loc, fc2w, fc2b,
                                       w.EMB + (long long)bag * 551, out + (long long)bag * 186);
  }
  // slide-level prediction (EMB holds 3*551 = 1653 contiguous)
  pred1_kernel<<<16, 256, 0, stream>>>(w.EMB, pw1, w.Y1);
  pred_lnrelu_kernel<<<1, 256, 0, stream>>>(w.Y1, pg1, pb1);
  pred2_kernel<<<1, 256, 0, stream>>>(w.Y1, pw2, pg2, pb2, out + 3 * 186);
}
